// MultiheadAttention_59846074302908
// MI455X (gfx1250) — matrix operand order repaired and checked by reading, not compiled, not run on hardware
//
#include <hip/hip_runtime.h>
#include <hip/hip_bf16.h>

#define NH 8
#define HD 64
#define EMB 512
#define SEQ 2048
#define BATCH 4
#define ROWS (BATCH * SEQ) /* 8192 */

typedef __attribute__((ext_vector_type(16))) _Float16 v16h;
typedef __attribute__((ext_vector_type(8))) float v8f;

static __device__ __forceinline__ v8f wmma_f16(v16h a, v16h b, v8f c) {
    // D = A(16x32 f16) * B(32x16 f16) + C(16x16 f32)
    return __builtin_amdgcn_wmma_f32_16x16x32_f16(false, a, false, b, (short)0, c,
                                                  false, false);
}

// ---- fragment loaders (ISA 7.12.2 layouts) ------------------------------
// A 16x32 f16: lane L -> row M = L&15; lanes 0-15 hold K 0..7 & 16..23,
// lanes 16-31 hold K 8..15 & 24..31 (two contiguous 8-half chunks).
static __device__ __forceinline__ v16h load_a_f16(const _Float16* base, int ld,
                                                  int m0, int k0, int lane) {
    const _Float16* row =
        base + (size_t)(m0 + (lane & 15)) * ld + k0 + ((lane >> 4) * 8);
    v16h a;
#pragma unroll
    for (int j = 0; j < 8; ++j) {
        a[j] = row[j];
        a[8 + j] = row[16 + j];
    }
    return a;
}
static __device__ __forceinline__ v16h load_a_f32(const float* base, int ld,
                                                  int m0, int k0, int lane) {
    const float* row =
        base + (size_t)(m0 + (lane & 15)) * ld + k0 + ((lane >> 4) * 8);
    v16h a;
#pragma unroll
    for (int j = 0; j < 8; ++j) {
        a[j] = (_Float16)row[j];
        a[8 + j] = (_Float16)row[16 + j];
    }
    return a;
}
// B 32x16: B[k][n] = src[n][k]; lane L -> col N = L&15, lanes 0-15 hold
// K 0..7 & 16..23, lanes 16-31 hold K 8..15 & 24..31.
static __device__ __forceinline__ v16h load_b_rows_f16(const _Float16* base,
                                                       int ld, int n0, int k0,
                                                       int lane) {
    const _Float16* row =
        base + (size_t)(n0 + (lane & 15)) * ld + k0 + ((lane >> 4) * 8);
    v16h b;
#pragma unroll
    for (int j = 0; j < 8; ++j) {
        b[j] = row[j];
        b[8 + j] = row[16 + j];
    }
    return b;
}

// ---- kernel 0: one-time f32 -> f16 weight conversion ---------------------
__global__ void __launch_bounds__(256) cvtw_kernel(const float* __restrict__ src,
                                                   _Float16* __restrict__ dst,
                                                   int n) {
    const int i = blockIdx.x * 256 + threadIdx.x;
    if (i < n) dst[i] = (_Float16)src[i];
}

// ---- kernel 1: per-head input projection  Y = X_h @ W^T  (f32 -> f16) ----
__global__ void __launch_bounds__(256) proj_kernel(const float* __restrict__ X,
                                                   const _Float16* __restrict__ W,
                                                   _Float16* __restrict__ Y) {
    const int lane = threadIdx.x & 31;
    const int wave = threadIdx.x >> 5;
    const int m0 = (blockIdx.x * 8 + wave) * 16;  // 16 rows per wave
#pragma unroll 1
    for (int h = 0; h < NH; ++h) {
        const v16h a0 = load_a_f32(X, EMB, m0, h * HD, lane);
        const v16h a1 = load_a_f32(X, EMB, m0, h * HD + 32, lane);
#pragma unroll
        for (int nt = 0; nt < 4; ++nt) {
            const v16h b0 = load_b_rows_f16(W, HD, nt * 16, 0, lane);
            const v16h b1 = load_b_rows_f16(W, HD, nt * 16, 32, lane);
            v8f acc = {};
            acc = wmma_f16(a0, b0, acc);
            acc = wmma_f16(a1, b1, acc);
            const int e = nt * 16 + (lane & 15);
            const int g = lane >> 4;
#pragma unroll
            for (int r = 0; r < 8; ++r)
                Y[(size_t)(m0 + r + 8 * g) * EMB + h * HD + e] = (_Float16)acc[r];
        }
    }
}

// ---- kernel 2: fused flash attention per (b,h) ---------------------------
__global__ void __launch_bounds__(256) attn_kernel(const _Float16* __restrict__ Qp,
                                                   const _Float16* __restrict__ Kp,
                                                   const _Float16* __restrict__ Vp,
                                                   _Float16* __restrict__ Ao) {
    __shared__ _Float16 Kt[32 * 64];      // [key][dim]   (async-filled)
    __shared__ _Float16 Vt[64 * 32];      // [dim][key]   (transposed stage)
    __shared__ _Float16 Pl[8][16 * 32];   // per-wave P exchange (16 x 32)

    const int lane = threadIdx.x & 31;
    const int wave = threadIdx.x >> 5;
    const int b = blockIdx.y >> 3;
    const int h = blockIdx.y & 7;
    const _Float16* Qb = Qp + (size_t)b * SEQ * EMB + h * HD;
    const _Float16* Kb = Kp + (size_t)b * SEQ * EMB + h * HD;
    const _Float16* Vb = Vp + (size_t)b * SEQ * EMB + h * HD;
    const int m0 = blockIdx.x * 128 + wave * 16;  // q-row tile within S

    // Q A-fragments stay resident in VGPRs for the whole K loop
    const v16h qa0 = load_a_f16(Qb, EMB, m0, 0, lane);
    const v16h qa1 = load_a_f16(Qb, EMB, m0, 32, lane);

    v8f o0 = {}, o1 = {}, o2 = {}, o3 = {};
    float mrow[8], lrow[8];
#pragma unroll
    for (int r = 0; r < 8; ++r) {
        mrow[r] = -3.0e38f;
        lrow[r] = 0.0f;
    }

    // softmax done in exp2 domain: scale2 = log2(e)/sqrt(EMBED)
    const float scale2 = 0.06376737245816746f;
    const int lr = threadIdx.x >> 3;       // 0..31 key row for coop load
    const int lc = (threadIdx.x & 7) * 8;  // dim chunk (8 halves = 16 B)

    for (int s0 = 0; s0 < SEQ; s0 += 32) {
        __syncthreads();
        {
            const _Float16* ksrc = Kb + (size_t)(s0 + lr) * EMB + lc;
            const _Float16* vsrc = Vb + (size_t)(s0 + lr) * EMB + lc;
            // K tile: 16B/thread straight to LDS via async DMA path
            const unsigned kdst = (unsigned)(size_t)(void*)&Kt[lr * 64 + lc];
            asm volatile("global_load_async_to_lds_b128 %0, %1, off"
                         :: "v"(kdst), "v"(ksrc) : "memory");
            // V tile: transpose through VGPRs into LDS
#pragma unroll
            for (int j = 0; j < 8; ++j) Vt[(lc + j) * 32 + lr] = vsrc[j];
            if (s0 + 32 < SEQ) {  // global_prefetch_b8 of next tile
                __builtin_prefetch(ksrc + 32 * EMB, 0, 1);
                __builtin_prefetch(vsrc + 32 * EMB, 0, 1);
            }
            asm volatile("s_wait_asynccnt 0" ::: "memory");
        }
        __syncthreads();

        // energy tile 16x32 = Q(16x64) . K^T(64x32), two n-tiles x two k-steps
        v8f e0 = {}, e1 = {};
        e0 = wmma_f16(qa0, load_b_rows_f16(Kt, 64, 0, 0, lane), e0);
        e0 = wmma_f16(qa1, load_b_rows_f16(Kt, 64, 0, 32, lane), e0);
        e1 = wmma_f16(qa0, load_b_rows_f16(Kt, 64, 16, 0, lane), e1);
        e1 = wmma_f16(qa1, load_b_rows_f16(Kt, 64, 16, 32, lane), e1);

        // online softmax in C/D layout: component r <-> row r+8*(lane>=16),
        // columns striped over 16 lanes -> shfl_xor(1,2,4,8) row reductions
        _Float16* P = &Pl[wave][0];
        const int c = lane & 15, g = lane >> 4;
        float alph[8];
#pragma unroll
        for (int r = 0; r < 8; ++r) {
            const float s0v = e0[r] * scale2;   // log2-domain score
            const float s1v = e1[r] * scale2;
            float mx = fmaxf(s0v, s1v);
#pragma unroll
            for (int i = 1; i < 16; i <<= 1) mx = fmaxf(mx, __shfl_xor(mx, i, 32));
            const float mnew = fmaxf(mrow[r], mx);
            const float a = exp2f(mrow[r] - mnew);
            const float p0 = exp2f(s0v - mnew);
            const float p1 = exp2f(s1v - mnew);
            float rs = p0 + p1;
#pragma unroll
            for (int i = 1; i < 16; i <<= 1) rs += __shfl_xor(rs, i, 32);
            lrow[r] = lrow[r] * a + rs;
            mrow[r] = mnew;
            alph[r] = a;
            const int m = r + 8 * g;
            P[m * 32 + c] = (_Float16)p0;
            P[m * 32 + 16 + c] = (_Float16)p1;
        }
        // intra-wave LDS RAW fence (cross-lane redistribution of P)
        asm volatile("s_wait_dscnt 0" ::: "memory");

#pragma unroll
        for (int r = 0; r < 8; ++r) {
            o0[r] *= alph[r];
            o1[r] *= alph[r];
            o2[r] *= alph[r];
            o3[r] *= alph[r];
        }
        const v16h pa = load_a_f16(P, 32, 0, 0, lane);  // P as A 16x32
        o0 = wmma_f16(pa, load_b_rows_f16(Vt, 32, 0, 0, lane), o0);
        o1 = wmma_f16(pa, load_b_rows_f16(Vt, 32, 16, 0, lane), o1);
        o2 = wmma_f16(pa, load_b_rows_f16(Vt, 32, 32, 0, lane), o2);
        o3 = wmma_f16(pa, load_b_rows_f16(Vt, 32, 48, 0, lane), o3);
    }

    const int c = lane & 15, g = lane >> 4;
#pragma unroll
    for (int r = 0; r < 8; ++r) {
        const float inv = 1.0f / lrow[r];
        const size_t gr = (size_t)b * SEQ + m0 + r + 8 * g;
        _Float16* dst = Ao + gr * EMB + h * HD + c;
        dst[0] = (_Float16)(o0[r] * inv);
        dst[16] = (_Float16)(o1[r] * inv);
        dst[32] = (_Float16)(o2[r] * inv);
        dst[48] = (_Float16)(o3[r] * inv);
    }
}

// ---- kernel 3: output projection  Out = Ao @ Wo^T + bo -------------------
__global__ void __launch_bounds__(256) outproj_kernel(const _Float16* __restrict__ Ao,
                                                      const _Float16* __restrict__ Wo16,
                                                      const float* __restrict__ bo,
                                                      float* __restrict__ Out) {
    const int lane = threadIdx.x & 31;
    const int wave = threadIdx.x >> 5;
    const int m0 = blockIdx.x * 16;
    const int e0 = wave * 64;  // 8 waves cover N=512
    v8f acc[4] = {};
#pragma unroll 1
    for (int c0 = 0; c0 < EMB; c0 += 32) {
        const v16h a = load_a_f16(Ao, EMB, m0, c0, lane);
#pragma unroll
        for (int nt = 0; nt < 4; ++nt) {
            const v16h bf = load_b_rows_f16(Wo16, EMB, e0 + nt * 16, c0, lane);
            acc[nt] = wmma_f16(a, bf, acc[nt]);
        }
    }
    const int g = lane >> 4;
#pragma unroll
    for (int nt = 0; nt < 4; ++nt) {
        const int e = e0 + nt * 16 + (lane & 15);
        const float bias = bo[e];
#pragma unroll
        for (int r = 0; r < 8; ++r)
            Out[(size_t)(m0 + r + 8 * g) * EMB + e] = acc[nt][r] + bias;
    }
}

extern "C" void kernel_launch(void* const* d_in, const int* in_sizes, int n_in,
                              void* d_out, int out_size, void* d_ws, size_t ws_size,
                              hipStream_t stream) {
    (void)in_sizes; (void)n_in; (void)out_size; (void)ws_size;
    const float* values = (const float*)d_in[0];
    const float* keys   = (const float*)d_in[1];
    const float* query  = (const float*)d_in[2];
    const float* Wv     = (const float*)d_in[3];
    const float* Wk     = (const float*)d_in[4];
    const float* Wq     = (const float*)d_in[5];
    const float* Wo     = (const float*)d_in[6];
    const float* bo     = (const float*)d_in[7];
    float* out = (float*)d_out;

    const size_t tsz = (size_t)ROWS * EMB;  // halves per f16 tensor
    _Float16* Qp   = (_Float16*)d_ws;
    _Float16* Kp   = Qp + tsz;
    _Float16* Vp   = Kp + tsz;
    _Float16* Ao   = Vp + tsz;
    _Float16* Wo16 = Ao + tsz;
    _Float16* Wq16 = Wo16 + (size_t)EMB * EMB;
    _Float16* Wk16 = Wq16 + HD * HD;
    _Float16* Wv16 = Wk16 + HD * HD;  // ~32.5 MB total workspace

    cvtw_kernel<<<dim3((HD * HD + 255) / 256), 256, 0, stream>>>(Wq, Wq16, HD * HD);
    cvtw_kernel<<<dim3((HD * HD + 255) / 256), 256, 0, stream>>>(Wk, Wk16, HD * HD);
    cvtw_kernel<<<dim3((HD * HD + 255) / 256), 256, 0, stream>>>(Wv, Wv16, HD * HD);
    cvtw_kernel<<<dim3((EMB * EMB + 255) / 256), 256, 0, stream>>>(Wo, Wo16,
                                                                   EMB * EMB);

    proj_kernel<<<dim3(ROWS / 128), 256, 0, stream>>>(query, Wq16, Qp);
    proj_kernel<<<dim3(ROWS / 128), 256, 0, stream>>>(keys, Wk16, Kp);
    proj_kernel<<<dim3(ROWS / 128), 256, 0, stream>>>(values, Wv16, Vp);
    attn_kernel<<<dim3(SEQ / 128, BATCH * NH), 256, 0, stream>>>(Qp, Kp, Vp, Ao);
    outproj_kernel<<<dim3(ROWS / 16), 256, 0, stream>>>(Ao, Wo16, bo, out);
}
